// Encoder_43774306680891
// MI455X (gfx1250) — matrix operand order repaired and checked by reading, not compiled, not run on hardware
//
#include <hip/hip_runtime.h>
#include <hip/hip_bf16.h>

// ---------------------------------------------------------------------------
// CDNA5 (gfx1250) encoder pipeline, round 3.
//   - NHWC bf16 activations, channel-padded to 32 -> B-fragment = 2x b128 loads
//   - zero-padded weights/channels -> no tails, no clamps in the K loop
//   - 2 M-tiles per wave, bf16x8 vector stores
//   - v_wmma_f32_16x16x32_bf16 f32-accumulate; LDS LayerNorms; prefetch
// ---------------------------------------------------------------------------

#define NB 8
#define BNS_F 0.99999499998749993f   // 1/sqrt(1+1e-5)

typedef __attribute__((ext_vector_type(16))) __bf16   bf16x16;
typedef __attribute__((ext_vector_type(8)))  __bf16   bf16x8;
typedef __attribute__((ext_vector_type(8)))  float    floatx8;
typedef __attribute__((ext_vector_type(4)))  float    floatx4;
typedef __attribute__((ext_vector_type(4)))  unsigned uint4v;

__device__ __forceinline__ __bf16 f2bf(float f) {
    unsigned u = __builtin_bit_cast(unsigned, f);
    u += 0x7fffu + ((u >> 16) & 1u);                  // round-to-nearest-even
    unsigned short h = (unsigned short)(u >> 16);
    return __builtin_bit_cast(__bf16, h);
}
__device__ __forceinline__ float bf2f(__bf16 h) {
    unsigned u = (unsigned)__builtin_bit_cast(unsigned short, h) << 16;
    return __builtin_bit_cast(float, u);
}

// ---------------------------------------------------------------------------
// Pack f32 OIHW weights into zero-padded bf16 [CoutP][kk][CinP].
// ---------------------------------------------------------------------------
__global__ void enc_pack_w(const float* __restrict__ w, __bf16* __restrict__ o,
                           int Cout, int Cin, int kk, int CoutP, int CinP) {
    int idx = blockIdx.x * blockDim.x + threadIdx.x;
    int total = CoutP * kk * CinP;
    if (idx >= total) return;
    int ci = idx % CinP; int t = idx / CinP;
    int tap = t % kk;    int co = t / kk;
    float v = (co < Cout && ci < Cin)
                  ? w[(size_t)co * Cin * kk + (size_t)ci * kk + tap] : 0.f;
    o[idx] = f2bf(v);
}

// ---------------------------------------------------------------------------
// 7x7 stride-2 pad-3 conv, Cin=1 -> 48, relu(BNS*conv).  f32 NCHW out.
// ---------------------------------------------------------------------------
__global__ void enc_conv7s2(const float* __restrict__ img,
                            const float* __restrict__ w,
                            float* __restrict__ out) {
    const int HO = 128, WO = 512;
    int idx = blockIdx.x * blockDim.x + threadIdx.x;
    int total = NB * 48 * HO * WO;
    if (idx >= total) return;
    int x = idx % WO; int t = idx / WO;
    int y = t % HO;   t /= HO;
    int co = t % 48;  int b = t / 48;
    const float* wp = w + co * 49;
    const float* ip = img + (size_t)b * 256 * 1024;
    int iy0 = 2 * y - 3, ix0 = 2 * x - 3;
    float acc = 0.f;
    for (int ky = 0; ky < 7; ++ky) {
        int iy = iy0 + ky;
        if (iy < 0 || iy >= 256) continue;
        for (int kx = 0; kx < 7; ++kx) {
            int ix = ix0 + kx;
            if (ix < 0 || ix >= 1024) continue;
            acc += wp[ky * 7 + kx] * ip[(size_t)iy * 1024 + ix];
        }
    }
    acc *= BNS_F;
    out[idx] = acc > 0.f ? acc : 0.f;
}

// ---------------------------------------------------------------------------
// stem maxpool: f32 NCHW [8][48][128][512] -> bf16 NHWC [8][64*256][CoutStride]
// ---------------------------------------------------------------------------
__global__ void enc_pool_stem(const float* __restrict__ in, __bf16* __restrict__ out,
                              int CoutStride) {
    const int C = 48, Ho = 64, Wo = 256, Hi = 128, Wi = 512;
    int idx = blockIdx.x * blockDim.x + threadIdx.x;
    int total = NB * Ho * Wo * C;
    if (idx >= total) return;
    int c = idx % C; int t = idx / C;
    int x = t % Wo;  t /= Wo;
    int y = t % Ho;  int b = t / Ho;
    const float* ip = in + (size_t)b * C * Hi * Wi + (size_t)c * Hi * Wi
                         + (size_t)(2 * y) * Wi + 2 * x;
    float v = fmaxf(fmaxf(ip[0], ip[1]), fmaxf(ip[Wi], ip[Wi + 1]));
    out[((size_t)b * Ho * Wo + (size_t)y * Wo + x) * CoutStride + c] = f2bf(v);
}

// ---------------------------------------------------------------------------
// NHWC bf16 2x2 avg pool: [b][2Ho*2Wo][CinStride] -> [b][Ho*Wo][CoutStride]
// ---------------------------------------------------------------------------
__global__ void enc_pool_nhwc(const __bf16* __restrict__ in, __bf16* __restrict__ out,
                              int C, int Ho, int Wo, int CinStride, int CoutStride) {
    int idx = blockIdx.x * blockDim.x + threadIdx.x;
    int total = NB * Ho * Wo * C;
    if (idx >= total) return;
    int c = idx % C; int t = idx / C;
    int x = t % Wo;  t /= Wo;
    int y = t % Ho;  int b = t / Ho;
    int Wi = Wo * 2;
    const __bf16* ip = in + ((size_t)b * (Ho * 2) * Wi + (size_t)(2 * y) * Wi + 2 * x)
                              * CinStride + c;
    float v = (bf2f(ip[0]) + bf2f(ip[CinStride])
             + bf2f(ip[(size_t)Wi * CinStride])
             + bf2f(ip[(size_t)(Wi + 1) * CinStride])) * 0.25f;
    out[((size_t)b * Ho * Wo + (size_t)y * Wo + x) * CoutStride + c] = f2bf(v);
}

// ---------------------------------------------------------------------------
// conv-as-GEMM via WMMA. KH in {1,3}; F32OUT: 0 -> bf16 NHWC out, 1 -> f32 NHWC.
// in : bf16 NHWC [b][H*W][CinStride], channels >= Cin are zero.
// wp : bf16 [CoutP][kk][CinP], zero padded.
// One wave computes 32(Cout) x 16(pixels); 2 WMMAs per K-step share B.
// ---------------------------------------------------------------------------
template <int KH, int F32OUT>
__global__ void enc_conv_gemm_wmma(const __bf16* __restrict__ in, int CinStride,
                                   const __bf16* __restrict__ wp,
                                   const float* __restrict__ bias,
                                   void* __restrict__ outv, int CoutStride, int chOff,
                                   int CinP, int Cout,
                                   int H, int W, int relu, float scale) {
    const int lane = threadIdx.x & 31;
    const int wave = threadIdx.x >> 5;
    const int wTiles = W >> 4;
    const int nTiles = H * wTiles;
    const int nt = blockIdx.x * 8 + wave;
    if (nt >= nTiles) return;                 // wave-uniform
    const int b  = blockIdx.z;
    const int mt = blockIdx.y;                // 32-row super-tile
    const int y  = nt / wTiles;
    const int x0 = (nt - y * wTiles) << 4;
    const int m    = lane & 15;
    const int half = lane >> 4;
    const int kk   = KH * KH;
    const int pad  = (KH == 3) ? 1 : 0;
    const int HW   = H * W;

    const __bf16* inb = in + (size_t)b * HW * CinStride;
    const __bf16* w0  = wp + (size_t)(mt * 32 + m) * kk * CinP + 8 * half;
    const __bf16* w1  = w0 + (size_t)16 * kk * CinP;
    __builtin_prefetch(w0, 0, 0);             // global_prefetch_b8

    floatx8 acc0 = {0.f,0.f,0.f,0.f,0.f,0.f,0.f,0.f};
    floatx8 acc1 = {0.f,0.f,0.f,0.f,0.f,0.f,0.f,0.f};
    const uint4v uz = {0u, 0u, 0u, 0u};

    for (int ky = 0; ky < KH; ++ky) {
        const int iy  = y + ky - pad;
        const bool yok = (iy >= 0) && (iy < H);
        for (int kx = 0; kx < KH; ++kx) {
            const int ix  = x0 + m + kx - pad;
            const bool ok = yok && (ix >= 0) && (ix < W);
            const int iyc = yok ? iy : 0;
            const int ixc = (ix < 0) ? 0 : (ix >= W ? W - 1 : ix);
            const __bf16* bpix = inb + (size_t)(iyc * W + ixc) * CinStride + 8 * half;
            const __bf16* wt0  = w0 + (ky * KH + kx) * CinP;
            const __bf16* wt1  = w1 + (ky * KH + kx) * CinP;
            for (int k0 = 0; k0 < CinP; k0 += 32) {
                // A fragments: 4 x 16B unconditional loads
                bf16x8 a00 = *(const bf16x8*)(wt0 + k0);
                bf16x8 a01 = *(const bf16x8*)(wt0 + k0 + 16);
                bf16x8 a10 = *(const bf16x8*)(wt1 + k0);
                bf16x8 a11 = *(const bf16x8*)(wt1 + k0 + 16);
                // B fragment: 2 x 16B loads (same k order as A), spatial mask
                bf16x8 b0 = *(const bf16x8*)(bpix + k0);
                bf16x8 b1 = *(const bf16x8*)(bpix + k0 + 16);
                if (KH == 3) {
                    uint4v u0 = __builtin_bit_cast(uint4v, b0);
                    uint4v u1 = __builtin_bit_cast(uint4v, b1);
                    u0 = ok ? u0 : uz;
                    u1 = ok ? u1 : uz;
                    b0 = __builtin_bit_cast(bf16x8, u0);
                    b1 = __builtin_bit_cast(bf16x8, u1);
                }
                bf16x16 av0, av1, bv;
#pragma unroll
                for (int t = 0; t < 8; ++t) {
                    av0[t] = a00[t]; av0[8 + t] = a01[t];
                    av1[t] = a10[t]; av1[8 + t] = a11[t];
                    bv[t]  = b0[t];  bv[8 + t]  = b1[t];
                }
                acc0 = __builtin_amdgcn_wmma_f32_16x16x32_bf16(
                           false, av0, false, bv, (short)0, acc0, false, false);
                acc1 = __builtin_amdgcn_wmma_f32_16x16x32_bf16(
                           false, av1, false, bv, (short)0, acc1, false, false);
            }
        }
    }

    const size_t pixBase = ((size_t)b * HW + (size_t)y * W + x0 + m) * CoutStride + chOff;
#pragma unroll
    for (int t = 0; t < 2; ++t) {
        const floatx8& a = t ? acc1 : acc0;
        int rb = mt * 32 + t * 16 + 8 * half;     // first of 8 consecutive channels
        if (rb < Cout) {
            if (F32OUT) {
                float* ob = (float*)outv + pixBase + rb;
                floatx4 lo, hi;
#pragma unroll
                for (int r = 0; r < 4; ++r) {
                    float v0 = a[r] * scale, v1 = a[r + 4] * scale;
                    if (bias) { v0 += bias[rb + r]; v1 += bias[rb + r + 4]; }
                    if (relu) { v0 = v0 > 0.f ? v0 : 0.f; v1 = v1 > 0.f ? v1 : 0.f; }
                    lo[r] = v0; hi[r] = v1;
                }
                *(floatx4*)(ob)     = lo;
                *(floatx4*)(ob + 4) = hi;
            } else {
                bf16x8 ov;
#pragma unroll
                for (int r = 0; r < 8; ++r) {
                    float v = a[r] * scale;
                    if (bias) v += bias[rb + r];
                    if (relu) v = v > 0.f ? v : 0.f;
                    ov[r] = f2bf(v);
                }
                *(bf16x8*)((__bf16*)outv + pixBase + rb) = ov;
            }
        }
    }
}

// ---------------------------------------------------------------------------
// f32 -> bf16 mirror (projection nodes)
// ---------------------------------------------------------------------------
__global__ void enc_cvt_bf16(const float* __restrict__ in, __bf16* __restrict__ out,
                             int total) {
    int idx = blockIdx.x * blockDim.x + threadIdx.x;
    if (idx >= total) return;
    out[idx] = f2bf(in[idx]);
}

// ---------------------------------------------------------------------------
// Gram matrix G[b] = X X^T; xbf = bf16 nodes [b][1024][256] (row-major).
// A and B fragments are contiguous 16B loads.
// ---------------------------------------------------------------------------
__global__ void enc_gram_wmma(const __bf16* __restrict__ xbf, float* __restrict__ G) {
    const int lane = threadIdx.x & 31;
    const int wave = threadIdx.x >> 5;
    const int nt = blockIdx.x * 8 + wave;
    const int mt = blockIdx.y;
    const int b  = blockIdx.z;
    const int m    = lane & 15;
    const int half = lane >> 4;
    const __bf16* xb   = xbf + (size_t)b * 1024 * 256;
    const __bf16* arow = xb + (size_t)(mt * 16 + m) * 256 + 8 * half;
    const __bf16* brow = xb + (size_t)(nt * 16 + m) * 256 + 8 * half;
    floatx8 acc = {0.f,0.f,0.f,0.f,0.f,0.f,0.f,0.f};
#pragma unroll
    for (int k0 = 0; k0 < 256; k0 += 32) {
        bf16x8 a0 = *(const bf16x8*)(arow + k0);
        bf16x8 a1 = *(const bf16x8*)(arow + k0 + 16);
        bf16x8 b0 = *(const bf16x8*)(brow + k0);
        bf16x8 b1 = *(const bf16x8*)(brow + k0 + 16);
        bf16x16 av, bv;
#pragma unroll
        for (int t = 0; t < 8; ++t) {
            av[t] = a0[t]; av[8 + t] = a1[t];
            bv[t] = b0[t]; bv[8 + t] = b1[t];
        }
        acc = __builtin_amdgcn_wmma_f32_16x16x32_bf16(
                  false, av, false, bv, (short)0, acc, false, false);
    }
    float* g = G + (size_t)b * 1024 * 1024;
#pragma unroll
    for (int r = 0; r < 8; ++r) {
        int i = mt * 16 + r + 8 * half;
        g[(size_t)i * 1024 + nt * 16 + m] = acc[r];
    }
}

// ---------------------------------------------------------------------------
// top-8 smallest d2 per node
// ---------------------------------------------------------------------------
__global__ void enc_knn_topk(const float* __restrict__ G, int* __restrict__ nb) {
    int idx = blockIdx.x * blockDim.x + threadIdx.x;
    if (idx >= NB * 1024) return;
    int b = idx >> 10, i = idx & 1023;
    const float* g = G + (size_t)b * 1024 * 1024;
    float sqi = g[(size_t)i * 1024 + i];
    const float* gi = g + (size_t)i * 1024;
    float bd[8]; int bi[8];
#pragma unroll
    for (int k = 0; k < 8; ++k) { bd[k] = 3.0e38f; bi[k] = 0; }
    for (int j = 0; j < 1024; ++j) {
        float d2 = sqi + g[(size_t)j * 1024 + j] - 2.f * gi[j];
        if (j == i) d2 += 1.0e9f;
        if (d2 < bd[7]) {
            bd[7] = d2; bi[7] = j;
#pragma unroll
            for (int t = 6; t >= 0; --t) {
                if (bd[t + 1] < bd[t]) {
                    float tf = bd[t]; bd[t] = bd[t + 1]; bd[t + 1] = tf;
                    int   ti = bi[t]; bi[t] = bi[t + 1]; bi[t + 1] = ti;
                }
            }
        }
    }
    int* o = nb + (size_t)idx * 8;
#pragma unroll
    for (int k = 0; k < 8; ++k) o[k] = bi[k];
}

// ---------------------------------------------------------------------------
// graph aggregation + residual + LayerNorm; nodes f32 [b][1024][256]
// ---------------------------------------------------------------------------
__global__ void enc_graph_agg_ln(const float* __restrict__ xf,
                                 const int* __restrict__ nb,
                                 const float* __restrict__ gam,
                                 const float* __restrict__ bet,
                                 float* __restrict__ refined) {
    const int bp = blockIdx.x;
    const int b = bp >> 10, p = bp & 1023;
    const int c = threadIdx.x;
    const int py = p >> 6, px = p & 63;
    const float* xb = xf + (size_t)b * 1024 * 256;
    float agg = 0.f, cnt = 8.f;
    if (px > 0)  { agg += xb[(size_t)(p - 1)  * 256 + c]; cnt += 1.f; }
    if (px < 63) { agg += xb[(size_t)(p + 1)  * 256 + c]; cnt += 1.f; }
    if (py > 0)  { agg += xb[(size_t)(p - 64) * 256 + c]; cnt += 1.f; }
    if (py < 15) { agg += xb[(size_t)(p + 64) * 256 + c]; cnt += 1.f; }
    const int* nbi = nb + (size_t)bp * 8;
#pragma unroll
    for (int k = 0; k < 8; ++k) agg += xb[(size_t)nbi[k] * 256 + c];
    float r = agg / cnt + xb[(size_t)p * 256 + c];

    __shared__ float s1[256];
    __shared__ float s2[256];
    s1[c] = r; s2[c] = r * r;
    __syncthreads();
    for (int off = 128; off > 0; off >>= 1) {
        if (c < off) { s1[c] += s1[c + off]; s2[c] += s2[c + off]; }
        __syncthreads();
    }
    float mean = s1[0] * (1.f / 256.f);
    float var  = s2[0] * (1.f / 256.f) - mean * mean;
    float o = (r - mean) * rsqrtf(var + 1e-5f) * gam[c] + bet[c];
    refined[(size_t)bp * 256 + c] = o;
}

// ---------------------------------------------------------------------------
// pos-enc + final LayerNorm + outputs
// ---------------------------------------------------------------------------
__global__ void enc_posenc_ln_out(const float* __restrict__ refined,
                                  const float* __restrict__ mask,
                                  const float* __restrict__ gam,
                                  const float* __restrict__ bet,
                                  float* __restrict__ outx,
                                  float* __restrict__ outm) {
    const int bp = blockIdx.x;
    const int b = bp >> 10, p = bp & 1023;
    const int c = threadIdx.x;
    const int py = p >> 6, px = p & 63;
    float mval = mask[(size_t)b * 256 * 1024 + (size_t)(py * 16) * 1024 + px * 16];
    float pos = (c & 1) ? (px * (1.f / 63.f)) : (py * (1.f / 15.f));
    float v = refined[(size_t)bp * 256 + c] + pos * mval;

    __shared__ float s1[256];
    __shared__ float s2[256];
    s1[c] = v; s2[c] = v * v;
    __syncthreads();
    for (int off = 128; off > 0; off >>= 1) {
        if (c < off) { s1[c] += s1[c + off]; s2[c] += s2[c + off]; }
        __syncthreads();
    }
    float mean = s1[0] * (1.f / 256.f);
    float var  = s2[0] * (1.f / 256.f) - mean * mean;
    float o = (v - mean) * rsqrtf(var + 1e-5f) * gam[c] + bet[c];
    outx[(size_t)bp * 256 + c] = o;
    if (c == 0) outm[bp] = mval;
}

// ---------------------------------------------------------------------------
// host-side orchestration
// ---------------------------------------------------------------------------
extern "C" void kernel_launch(void* const* d_in, const int* in_sizes, int n_in,
                              void* d_out, int out_size, void* d_ws, size_t ws_size,
                              hipStream_t stream) {
    (void)in_sizes; (void)n_in; (void)out_size; (void)ws_size;

    const float* img     = (const float*)d_in[0];
    const float* imgmask = (const float*)d_in[1];
    const float* conv1_w = (const float*)d_in[2];
    const float* t1_w   = (const float*)d_in[35];
    const float* t2_w   = (const float*)d_in[68];
    const float* proj_w = (const float*)d_in[101];
    const float* proj_b = (const float*)d_in[102];
    const float* gn_g   = (const float*)d_in[103];
    const float* gn_b   = (const float*)d_in[104];
    const float* ln_g   = (const float*)d_in[105];
    const float* ln_b   = (const float*)d_in[106];

    // channel-padded NHWC strides
    const int S1 = 448, S2 = 608, S3 = 704;          // blk1/2/3 (432/600/684 -> pad32)
    const int T1 = 224, T2 = 320;                    // transition outputs (216/300)
    const int P1 = 64 * 256, P2 = 32 * 128, P3 = 16 * 64;

    char* wsb = (char*)d_ws;
    size_t off = 0;
    auto carve = [&](size_t bytes) { char* p = wsb + off; off += bytes; return p; };
    float*  bufAf = (float*)carve(100663296);        // conv1 f32 / t-outs / gram
    __bf16* blk1  = (__bf16*)carve((size_t)NB * P1 * S1 * 2);   // 117,440,512 B
    __bf16* hbuf  = (__bf16*)carve((size_t)NB * P1 * 96 * 2);   //  25,165,824 B
    __bf16* blk2  = (__bf16*)carve((size_t)NB * P2 * S2 * 2);   //  39,845,888 B
    __bf16* blk3  = (__bf16*)carve((size_t)NB * P3 * S3 * 2);   //  11,534,336 B
    float*  projf = (float*)carve((size_t)NB * P3 * 256 * 4);   //   8,388,608 B
    __bf16* projh = (__bf16*)carve((size_t)NB * P3 * 256 * 2);  //   4,194,304 B
    __bf16* wpack = (__bf16*)carve(524288);                     // packed weights
    __bf16* tbufA = (__bf16*)bufAf;                  // bf16 view for transitions
    float*  gram    = bufAf;                         // 33.5 MB
    float*  refined = bufAf + 8388608;               //  8.4 MB
    int*    nbuf    = (int*)(bufAf + 8388608 + 2097152);

    auto gemm = [&](const __bf16* in, int CinStride, const float* w,
                    const float* bias, void* out, int CoutStride, int chOff,
                    int Cin, int Cout, int H, int W, int kh, int relu,
                    float scale, int f32out) {
        int kk    = kh * kh;
        int CinP  = (Cin + 31) & ~31;
        int CoutP = (Cout + 31) & ~31;
        int pt = CoutP * kk * CinP;
        enc_pack_w<<<(pt + 255) / 256, 256, 0, stream>>>(w, wpack, Cout, Cin, kk,
                                                         CoutP, CinP);
        int nT = H * (W / 16);
        dim3 grid((nT + 7) / 8, CoutP / 32, NB);
        if (kh == 1 && !f32out)
            enc_conv_gemm_wmma<1,0><<<grid, 256, 0, stream>>>(in, CinStride, wpack,
                bias, out, CoutStride, chOff, CinP, Cout, H, W, relu, scale);
        else if (kh == 3)
            enc_conv_gemm_wmma<3,0><<<grid, 256, 0, stream>>>(in, CinStride, wpack,
                bias, out, CoutStride, chOff, CinP, Cout, H, W, relu, scale);
        else
            enc_conv_gemm_wmma<1,1><<<grid, 256, 0, stream>>>(in, CinStride, wpack,
                bias, out, CoutStride, chOff, CinP, Cout, H, W, relu, scale);
    };

    // ---- zero concat buffers (padding channels must be 0 for tail-free K) ----
    hipMemsetAsync(blk1, 0, (size_t)NB * P1 * S1 * 2, stream);
    hipMemsetAsync(blk2, 0, (size_t)NB * P2 * S2 * 2, stream);
    hipMemsetAsync(blk3, 0, (size_t)NB * P3 * S3 * 2, stream);

    // ---- stem ----
    {
        int total = NB * 48 * 128 * 512;
        enc_conv7s2<<<(total + 255) / 256, 256, 0, stream>>>(img, conv1_w, bufAf);
        int tp = NB * P1 * 48;
        enc_pool_stem<<<(tp + 255) / 256, 256, 0, stream>>>(bufAf, blk1, S1);
    }

    // ---- dense block 1 (64x256) ----
    for (int i = 0; i < 16; ++i) {
        int Cin = 48 + 24 * i;
        gemm(blk1, S1, (const float*)d_in[3 + i],  nullptr, hbuf, 96, 0,
             Cin, 96, 64, 256, 1, 1, BNS_F, 0);
        gemm(hbuf, 96, (const float*)d_in[19 + i], nullptr, blk1, S1, Cin,
             96, 24, 64, 256, 3, 1, BNS_F, 0);
    }
    // transition 1
    hipMemsetAsync(tbufA, 0, (size_t)NB * P1 * T1 * 2, stream);
    gemm(blk1, S1, t1_w, nullptr, tbufA, T1, 0, 432, 216, 64, 256, 1, 1, BNS_F, 0);
    {
        int tp = NB * P2 * 216;
        enc_pool_nhwc<<<(tp + 255) / 256, 256, 0, stream>>>(tbufA, blk2, 216, 32, 128,
                                                            T1, S2);
    }

    // ---- dense block 2 (32x128) ----
    for (int i = 0; i < 16; ++i) {
        int Cin = 216 + 24 * i;
        gemm(blk2, S2, (const float*)d_in[36 + i], nullptr, hbuf, 96, 0,
             Cin, 96, 32, 128, 1, 1, BNS_F, 0);
        gemm(hbuf, 96, (const float*)d_in[52 + i], nullptr, blk2, S2, Cin,
             96, 24, 32, 128, 3, 1, BNS_F, 0);
    }
    // transition 2
    hipMemsetAsync(tbufA, 0, (size_t)NB * P2 * T2 * 2, stream);
    gemm(blk2, S2, t2_w, nullptr, tbufA, T2, 0, 600, 300, 32, 128, 1, 1, BNS_F, 0);
    {
        int tp = NB * P3 * 300;
        enc_pool_nhwc<<<(tp + 255) / 256, 256, 0, stream>>>(tbufA, blk3, 300, 16, 64,
                                                            T2, S3);
    }

    // ---- dense block 3 (16x64) ----
    for (int i = 0; i < 16; ++i) {
        int Cin = 300 + 24 * i;
        gemm(blk3, S3, (const float*)d_in[69 + i], nullptr, hbuf, 96, 0,
             Cin, 96, 16, 64, 1, 1, BNS_F, 0);
        gemm(hbuf, 96, (const float*)d_in[85 + i], nullptr, blk3, S3, Cin,
             96, 24, 16, 64, 3, 1, BNS_F, 0);
    }

    // ---- projection (f32 out) + bf16 mirror ----
    gemm(blk3, S3, proj_w, proj_b, projf, 256, 0, 684, 256, 16, 64, 1, 0, BNS_F, 1);
    {
        int total = NB * P3 * 256;
        enc_cvt_bf16<<<(total + 255) / 256, 256, 0, stream>>>(projf, projh, total);
    }

    // ---- graph refine ----
    enc_gram_wmma<<<dim3(8, 64, NB), 256, 0, stream>>>(projh, gram);
    enc_knn_topk<<<(NB * 1024 + 255) / 256, 256, 0, stream>>>(gram, nbuf);
    enc_graph_agg_ln<<<NB * 1024, 256, 0, stream>>>(projf, nbuf, gn_g, gn_b, refined);

    // ---- pos-enc + final LN + outputs ----
    float* outx = (float*)d_out;
    float* outm = outx + (size_t)NB * 16 * 64 * 256;
    enc_posenc_ln_out<<<NB * 1024, 256, 0, stream>>>(refined, imgmask, ln_g, ln_b,
                                                     outx, outm);
}
